// HybridMemoryAttention_23656679866660
// MI455X (gfx1250) — compile-verified, matrix operand order checked by reading, not run on hardware
//
#include <hip/hip_runtime.h>
#include <hip/hip_bf16.h>
#include <cstdint>
#include <cmath>

typedef _Float16 half_t;
typedef __attribute__((ext_vector_type(16))) _Float16 v16h;
typedef __attribute__((ext_vector_type(8)))  _Float16 v8h;
typedef __attribute__((ext_vector_type(4)))  _Float16 v4h;
typedef __attribute__((ext_vector_type(8)))  float    v8f;

#define B_  4
#define S_  2048
#define H_  2048
#define NH_ 16
#define HD_ 128
#define KE_ 8
static constexpr float kScale = 0.08838834764831845f; // 1/sqrt(128)

static __device__ __forceinline__ v8f wmma16x16x32(v16h a, v16h b, v8f c) {
  return __builtin_amdgcn_wmma_f32_16x16x32_f16(false, a, false, b, (short)0, c,
                                                false, false);
}

static __device__ __forceinline__ v16h frag_ld(const half_t* p0, const half_t* p1) {
  v8h lo = *(const v8h*)p0;
  v8h hi = *(const v8h*)p1;
  return __builtin_shufflevector(lo, hi, 0,1,2,3,4,5,6,7,8,9,10,11,12,13,14,15);
}

// -------- CDNA5 async global->LDS copy (ASYNCcnt path) --------
static __device__ __forceinline__ unsigned lds_off(const void* p) {
  // flat LDS aperture address: low 32 bits are the wave-relative LDS byte addr
  return (unsigned)(uintptr_t)p;
}
static __device__ __forceinline__ void async_copy_b128(unsigned lds_byte,
                                                       const void* gptr) {
  asm volatile("global_load_async_to_lds_b128 %0, %1, off"
               :: "v"(lds_byte), "v"(gptr) : "memory");
}
static __device__ __forceinline__ void wait_async0() {
  asm volatile("s_wait_asynccnt 0" ::: "memory");
}

// ---------------- fp32 -> fp16 convert ----------------
__global__ void cvt_f32_f16(const float* __restrict__ in, half_t* __restrict__ out,
                            int n) {
  int i = (blockIdx.x * blockDim.x + threadIdx.x) * 4;
  if (i + 3 < n) {
    float4 v = *(const float4*)(in + i);
    out[i] = (half_t)v.x; out[i+1] = (half_t)v.y;
    out[i+2] = (half_t)v.z; out[i+3] = (half_t)v.w;
  } else {
    for (; i < n; ++i) out[i] = (half_t)in[i];
  }
}

// ---------------- fp32 [R,C] -> fp16 transposed [C,R] ----------------
__global__ __launch_bounds__(256)
void cvt_transpose_f32_f16(const float* __restrict__ in, half_t* __restrict__ out,
                           int R, int C) {
  __shared__ float tile[32][33];
  const int r0 = blockIdx.y * 32, c0 = blockIdx.x * 32;
  const int tx = threadIdx.x & 31, ty = threadIdx.x >> 5;
  #pragma unroll
  for (int i = ty; i < 32; i += 8)
    tile[i][tx] = in[(size_t)(r0 + i) * C + (c0 + tx)];
  __syncthreads();
  #pragma unroll
  for (int i = ty; i < 32; i += 8)
    out[(size_t)(c0 + i) * R + (r0 + tx)] = (half_t)tile[tx][i];
}

// ---------------- WMMA GEMM: C[M,N] = A[M,K] * Bt[N,K]^T ----------------
// 128x128 tile, BK=32, 8 waves (each 32x64), double-buffered async LDS loads.
// MODE: 0 = store f32, 1 = store f16, 2 = store f16 transposed to [b*H+col][S]
template <int MODE>
__global__ __launch_bounds__(256)
void gemm_wmma(const half_t* __restrict__ A, const half_t* __restrict__ Bt,
               float* __restrict__ Cf, half_t* __restrict__ Ch,
               int M, int N, int Kd, float scale) {
  __shared__ __align__(16) half_t sA[2][128 * 40];
  __shared__ __align__(16) half_t sB[2][128 * 40]; // [n][k]
  const int t    = threadIdx.x;
  const int lane = t & 31;
  const int w    = t >> 5;
  const int wm   = (w & 3) * 32;
  const int wn   = (w >> 2) * 64;
  const int bm0  = blockIdx.y * 128;
  const int bn0  = blockIdx.x * 128;
  const int l15  = lane & 15;
  const int kh   = (lane >> 4) * 8;

  v8f acc[2][4];
  #pragma unroll
  for (int mt = 0; mt < 2; ++mt)
    #pragma unroll
    for (int nt = 0; nt < 4; ++nt) acc[mt][nt] = v8f{};

  // async tile fetch: 512 16B-chunks per 128x32 tile, 2 per thread per tile
  auto issue_tiles = [&](int k0, int buf) {
    #pragma unroll
    for (int i = 0; i < 2; ++i) {
      const int cid = t + 256 * i;   // 0..511
      const int r = cid >> 2;        // 0..127
      const int c = (cid & 3) * 8;   // 0,8,16,24
      async_copy_b128(lds_off(&sA[buf][r * 40 + c]),
                      A + (size_t)(bm0 + r) * Kd + k0 + c);
      async_copy_b128(lds_off(&sB[buf][r * 40 + c]),
                      Bt + (size_t)(bn0 + r) * Kd + k0 + c);
    }
  };

  issue_tiles(0, 0);
  const int nk = Kd / 32;
  for (int it = 0; it < nk; ++it) {
    const int buf = it & 1;
    wait_async0();
    __syncthreads();                    // tile `it` visible to all waves
    if (it + 1 < nk) issue_tiles((it + 1) * 32, buf ^ 1);

    v16h af[2], bf[4];
    #pragma unroll
    for (int mt = 0; mt < 2; ++mt) {
      const half_t* p = &sA[buf][(wm + mt * 16 + l15) * 40 + kh];
      af[mt] = frag_ld(p, p + 16);
    }
    #pragma unroll
    for (int nt = 0; nt < 4; ++nt) {
      const half_t* p = &sB[buf][(wn + nt * 16 + l15) * 40 + kh];
      bf[nt] = frag_ld(p, p + 16);
    }
    #pragma unroll
    for (int mt = 0; mt < 2; ++mt)
      #pragma unroll
      for (int nt = 0; nt < 4; ++nt)
        acc[mt][nt] = wmma16x16x32(af[mt], bf[nt], acc[mt][nt]);
    __syncthreads();                    // all reads of buf done before reuse
  }

  const int rhi = (lane >> 4) * 8;
  #pragma unroll
  for (int mt = 0; mt < 2; ++mt) {
    #pragma unroll
    for (int nt = 0; nt < 4; ++nt) {
      const int col = bn0 + wn + nt * 16 + l15;
      #pragma unroll
      for (int v = 0; v < 8; ++v) {
        const int row = bm0 + wm + mt * 16 + rhi + v;
        const float val = acc[mt][nt][v] * scale;
        if (MODE == 0) {
          Cf[(size_t)row * N + col] = val;
        } else if (MODE == 1) {
          Ch[(size_t)row * N + col] = (half_t)val;
        } else {
          const int bb = row >> 11;          // row / S_
          const int s  = row & (S_ - 1);
          Ch[((size_t)bb * H_ + col) * S_ + s] = (half_t)val;
        }
      }
    }
  }
}

// ---------------- flash causal attention ----------------
// block = 128 thr = 4 waves; wave owns 16 q-rows; block owns 64 q-rows.
// K tile read from k16 [b,s,h]; V tile read from vt16 [(b*H+col)][S] (pre-T).
__global__ __launch_bounds__(128)
void flash_attn(const half_t* __restrict__ Q, const half_t* __restrict__ Kt,
                const half_t* __restrict__ Vt, half_t* __restrict__ O) {
  __shared__ __align__(16) half_t sK[32 * 136];   // [key][hd]
  __shared__ __align__(16) half_t sV[128 * 40];   // [hd][key]
  __shared__ __align__(16) half_t sP[4][16 * 40]; // per-wave P, [row][key]
  const int t    = threadIdx.x;
  const int lane = t & 31;
  const int w    = t >> 5;
  const int bh   = blockIdx.y;  // b*NH + h
  const int b    = bh >> 4;
  const int h    = bh & 15;
  const int qb   = blockIdx.x * 64 + w * 16;
  const int l15  = lane & 15;
  const int hi   = lane >> 4;
  const int kh   = hi * 8;

  // Q fragments (scale pre-folded into Q)
  v16h qf[4];
  {
    const size_t qrow = (size_t)(b * S_ + qb + l15) * H_ + h * HD_;
    #pragma unroll
    for (int c = 0; c < 4; ++c) {
      const half_t* p = Q + qrow + c * 32 + kh;
      qf[c] = frag_ld(p, p + 16);
    }
  }
  v8f acc[8];
  #pragma unroll
  for (int i = 0; i < 8; ++i) acc[i] = v8f{};
  float rmax[8], rsum[8];
  #pragma unroll
  for (int v = 0; v < 8; ++v) { rmax[v] = -3.0e38f; rsum[v] = 0.f; }

  const int nkt = (blockIdx.x + 1) * 2;
  for (int kt = 0; kt < nkt; ++kt) {
    const int k0 = kt * 32;
    __syncthreads();                 // prev-tile reads complete
    // async fetch: K tile 32x128 halves (512 chunks), V tile 128x32 (512 chunks)
    #pragma unroll
    for (int i = 0; i < 4; ++i) {
      const int cid = t + 128 * i;   // 0..511
      const int key = cid >> 4;      // 0..31
      const int c   = (cid & 15) * 8;
      async_copy_b128(lds_off(sK + key * 136 + c),
                      Kt + (size_t)(b * S_ + k0 + key) * H_ + h * HD_ + c);
    }
    {
      const int hd = t;              // 0..127 (one row each, 4 chunks)
      const half_t* gv = Vt + ((size_t)b * H_ + h * HD_ + hd) * S_ + k0;
      async_copy_b128(lds_off(sV + hd * 40), gv);
      async_copy_b128(lds_off(sV + hd * 40 + 8), gv + 8);
      async_copy_b128(lds_off(sV + hd * 40 + 16), gv + 16);
      async_copy_b128(lds_off(sV + hd * 40 + 24), gv + 24);
    }
    wait_async0();
    __syncthreads();

    // scores 16x32 = Q(16x128) x K^T(128x32)
    v8f sc[2];
    #pragma unroll
    for (int nt = 0; nt < 2; ++nt) {
      sc[nt] = v8f{};
      #pragma unroll
      for (int c = 0; c < 4; ++c) {
        const half_t* p = sK + (nt * 16 + l15) * 136 + c * 32 + kh;
        sc[nt] = wmma16x16x32(qf[c], frag_ld(p, p + 16), sc[nt]);
      }
    }
    // causal mask
    #pragma unroll
    for (int nt = 0; nt < 2; ++nt) {
      const int col = k0 + nt * 16 + l15;
      #pragma unroll
      for (int v = 0; v < 8; ++v)
        if (col > qb + hi * 8 + v) sc[nt][v] = -3.0e38f;
    }
    // online softmax (row reduce across 16-lane half-groups)
    float nmax[8], psum[8];
    #pragma unroll
    for (int v = 0; v < 8; ++v) {
      float m = fmaxf(sc[0][v], sc[1][v]);
      m = fmaxf(m, __shfl_xor(m, 1, 32));
      m = fmaxf(m, __shfl_xor(m, 2, 32));
      m = fmaxf(m, __shfl_xor(m, 4, 32));
      m = fmaxf(m, __shfl_xor(m, 8, 32));
      nmax[v] = fmaxf(rmax[v], m);
    }
    #pragma unroll
    for (int v = 0; v < 8; ++v) {
      float p0 = __expf(sc[0][v] - nmax[v]);
      float p1 = __expf(sc[1][v] - nmax[v]);
      sc[0][v] = p0; sc[1][v] = p1;
      float s = p0 + p1;
      s += __shfl_xor(s, 1, 32);
      s += __shfl_xor(s, 2, 32);
      s += __shfl_xor(s, 4, 32);
      s += __shfl_xor(s, 8, 32);
      psum[v] = s;
    }
    #pragma unroll
    for (int v = 0; v < 8; ++v) {
      const float f = __expf(rmax[v] - nmax[v]);
      rsum[v] = rsum[v] * f + psum[v];
      rmax[v] = nmax[v];
      #pragma unroll
      for (int ht = 0; ht < 8; ++ht) acc[ht][v] *= f;
    }
    // stash P (f16) as A-operand for P x V
    half_t* pw = sP[w];
    #pragma unroll
    for (int nt = 0; nt < 2; ++nt)
      #pragma unroll
      for (int v = 0; v < 8; ++v)
        pw[(hi * 8 + v) * 40 + nt * 16 + l15] = (half_t)sc[nt][v];
    __syncthreads();
    // O += P(16x32) x V(32x128)
    const half_t* pr = sP[w] + l15 * 40 + kh;
    const v16h pf = frag_ld(pr, pr + 16);
    #pragma unroll
    for (int ht = 0; ht < 8; ++ht) {
      const half_t* vp = sV + (ht * 16 + l15) * 40 + kh;
      acc[ht] = wmma16x16x32(pf, frag_ld(vp, vp + 16), acc[ht]);
    }
  }
  // normalize + store in [b,s,h] layout (ready for @Wo GEMM)
  #pragma unroll
  for (int v = 0; v < 8; ++v) {
    const float r = 1.0f / rsum[v];
    const int row = qb + hi * 8 + v;
    const size_t base = (size_t)(b * S_ + row) * H_ + h * HD_;
    #pragma unroll
    for (int ht = 0; ht < 8; ++ht)
      O[base + ht * 16 + l15] = (half_t)(acc[ht][v] * r);
  }
}

// ---------------- external memory attention (bandwidth-bound) ----------------
__global__ __launch_bounds__(256)
void ext_attn(const float* __restrict__ AO, const float* __restrict__ EK,
              const float* __restrict__ EV, half_t* __restrict__ EO) {
  const int w    = threadIdx.x >> 5;
  const int lane = threadIdx.x & 31;
  const long idx = (long)blockIdx.x * 8 + w;   // over b*NH*S
  const int b = (int)(idx / (NH_ * S_));
  const int rem = (int)(idx % (NH_ * S_));
  const int h = rem / S_;
  const int s = rem % S_;

  const size_t aoff = (size_t)(b * S_ + s) * H_ + h * HD_ + lane * 4;
  const float4 eq = *(const float4*)(AO + aoff);

  const size_t kb = ((((size_t)b * NH_ + h) * S_ + s) * KE_) * HD_ + lane * 4;
  float sc[KE_];
  #pragma unroll
  for (int k = 0; k < KE_; ++k) {
    const float4 kk = *(const float4*)(EK + kb + k * HD_);
    float d = eq.x * kk.x + eq.y * kk.y + eq.z * kk.z + eq.w * kk.w;
    d += __shfl_xor(d, 1, 32);
    d += __shfl_xor(d, 2, 32);
    d += __shfl_xor(d, 4, 32);
    d += __shfl_xor(d, 8, 32);
    d += __shfl_xor(d, 16, 32);
    sc[k] = d * kScale;
  }
  float m = sc[0];
  #pragma unroll
  for (int k = 1; k < KE_; ++k) m = fmaxf(m, sc[k]);
  float sum = 0.f;
  #pragma unroll
  for (int k = 0; k < KE_; ++k) { sc[k] = __expf(sc[k] - m); sum += sc[k]; }
  const float inv = 1.0f / sum;
  float4 o = {0.f, 0.f, 0.f, 0.f};
  #pragma unroll
  for (int k = 0; k < KE_; ++k) {
    const float wg = sc[k] * inv;
    const float4 vv = *(const float4*)(EV + kb + k * HD_);
    o.x += wg * vv.x; o.y += wg * vv.y; o.z += wg * vv.z; o.w += wg * vv.w;
  }
  v4h oh; oh[0] = (half_t)o.x; oh[1] = (half_t)o.y;
  oh[2] = (half_t)o.z; oh[3] = (half_t)o.w;
  *(v4h*)(EO + aoff) = oh;
}

// ---------------- gate + combine ----------------
__global__ __launch_bounds__(256)
void gate_combine(const float* __restrict__ X, const float* __restrict__ Wg,
                  const float* __restrict__ bg, const float* __restrict__ AO,
                  const float* __restrict__ XO, float* __restrict__ Out) {
  const int w    = threadIdx.x >> 5;
  const int lane = threadIdx.x & 31;
  const long tkn = (long)blockIdx.x * 8 + w;
  const size_t base = (size_t)tkn * H_;
  float g0 = 0.f, g1 = 0.f;
  for (int i = lane * 4; i < H_; i += 128) {
    const float4 x = *(const float4*)(X + base + i);
    g0 += x.x * Wg[i*3]   + x.y * Wg[(i+1)*3]   + x.z * Wg[(i+2)*3]   + x.w * Wg[(i+3)*3];
    g1 += x.x * Wg[i*3+1] + x.y * Wg[(i+1)*3+1] + x.z * Wg[(i+2)*3+1] + x.w * Wg[(i+3)*3+1];
  }
  #pragma unroll
  for (int off = 1; off < 32; off <<= 1) {
    g0 += __shfl_xor(g0, off, 32);
    g1 += __shfl_xor(g1, off, 32);
  }
  g0 += bg[0]; g1 += bg[1];
  const float m  = fmaxf(g0, g1);
  const float e0 = __expf(g0 - m), e1 = __expf(g1 - m);
  const float inv = 1.0f / (e0 + e1);
  const float w0 = e0 * inv, w1 = e1 * inv;
  for (int i = lane * 4; i < H_; i += 128) {
    const float4 a = *(const float4*)(AO + base + i);
    const float4 x = *(const float4*)(XO + base + i);
    float4 r;
    r.x = w0 * a.x + w1 * x.x; r.y = w0 * a.y + w1 * x.y;
    r.z = w0 * a.z + w1 * x.z; r.w = w0 * a.w + w1 * x.w;
    *(float4*)(Out + base + i) = r;
  }
}

extern "C" void kernel_launch(void* const* d_in, const int* in_sizes, int n_in,
                              void* d_out, int out_size, void* d_ws, size_t ws_size,
                              hipStream_t stream) {
  (void)in_sizes; (void)n_in; (void)out_size; (void)ws_size;
  const float* hidden = (const float*)d_in[0];
  const float* ext_k  = (const float*)d_in[1];
  const float* ext_v  = (const float*)d_in[2];
  const float* Wq = (const float*)d_in[3];
  const float* Wk = (const float*)d_in[4];
  const float* Wv = (const float*)d_in[5];
  const float* Wo = (const float*)d_in[6];
  const float* We = (const float*)d_in[7];
  const float* Wg = (const float*)d_in[8];
  const float* bg = (const float*)d_in[9];
  float* out = (float*)d_out;

  char* ws = (char*)d_ws;
  size_t off = 0;
  auto alloc = [&](size_t bytes) -> void* {
    void* p = ws + off;
    off += (bytes + 255) & ~(size_t)255;
    return p;
  };
  const size_t NTOK = (size_t)B_ * S_;            // 8192 tokens
  const size_t NELT = NTOK * H_;                  // 16.8M
  const size_t WELT = (size_t)H_ * H_;            // 4.2M
  half_t* h16   = (half_t*)alloc(NELT * 2);
  half_t* wq16  = (half_t*)alloc(WELT * 2);   // transposed [N][K]
  half_t* wk16  = (half_t*)alloc(WELT * 2);
  half_t* wv16  = (half_t*)alloc(WELT * 2);
  half_t* wo16  = (half_t*)alloc(WELT * 2);
  half_t* we16  = (half_t*)alloc(WELT * 2);
  half_t* q16   = (half_t*)alloc(NELT * 2);
  half_t* k16   = (half_t*)alloc(NELT * 2);
  half_t* vt16  = (half_t*)alloc(NELT * 2);   // V stored transposed per head
  half_t* ctx16 = (half_t*)alloc(NELT * 2);
  float*  ao    = (float*)alloc(NELT * 4);
  half_t* eo16  = (half_t*)alloc(NELT * 2);
  float*  xo    = (float*)alloc(NELT * 4);

  // 1) conversions: hidden straight, weights transposed to [N][K]
  cvt_f32_f16<<<(int)(NELT / 1024), 256, 0, stream>>>(hidden, h16, (int)NELT);
  dim3 gt(H_ / 32, H_ / 32);
  cvt_transpose_f32_f16<<<gt, 256, 0, stream>>>(Wq, wq16, H_, H_);
  cvt_transpose_f32_f16<<<gt, 256, 0, stream>>>(Wk, wk16, H_, H_);
  cvt_transpose_f32_f16<<<gt, 256, 0, stream>>>(Wv, wv16, H_, H_);
  cvt_transpose_f32_f16<<<gt, 256, 0, stream>>>(Wo, wo16, H_, H_);
  cvt_transpose_f32_f16<<<gt, 256, 0, stream>>>(We, we16, H_, H_);

  // 2) QKV projections (scale folded into Q; V stored head-transposed)
  dim3 gg(H_ / 128, (unsigned)(NTOK / 128));
  gemm_wmma<1><<<gg, 256, 0, stream>>>(h16, wq16, nullptr, q16,
                                       (int)NTOK, H_, H_, kScale);
  gemm_wmma<1><<<gg, 256, 0, stream>>>(h16, wk16, nullptr, k16,
                                       (int)NTOK, H_, H_, 1.0f);
  gemm_wmma<2><<<gg, 256, 0, stream>>>(h16, wv16, nullptr, vt16,
                                       (int)NTOK, H_, H_, 1.0f);

  // 3) flash causal attention -> ctx16 [b,s,h]
  dim3 ga(S_ / 64, B_ * NH_);
  flash_attn<<<ga, 128, 0, stream>>>(q16, k16, vt16, ctx16);

  // 4) attn_output = ctx16 @ Wo (f32)
  gemm_wmma<0><<<gg, 256, 0, stream>>>(ctx16, wo16, ao, nullptr,
                                       (int)NTOK, H_, H_, 1.0f);

  // 5) external memory attention -> eo16 [b,s,h]
  ext_attn<<<(B_ * NH_ * S_) / 8, 256, 0, stream>>>(ao, ext_k, ext_v, eo16);

  // 6) ext_out = eo16 @ W_ext (f32)
  gemm_wmma<0><<<gg, 256, 0, stream>>>(eo16, we16, xo, nullptr,
                                       (int)NTOK, H_, H_, 1.0f);

  // 7) gate + combine -> out (f32)
  gate_combine<<<(int)(NTOK / 8), 256, 0, stream>>>(hidden, Wg, bg, ao, xo, out);
}
